// BaseObjectDetector_30674656428434
// MI455X (gfx1250) — matrix operand order repaired and checked by reading, not compiled, not run on hardware
//
#include <hip/hip_runtime.h>
#include <hip/hip_bf16.h>
#include <math.h>

// Problem constants (match reference)
#define BDIM    16
#define NPRED   25200
#define NCLS    80
#define MAX_DET 300
#define CONF_T  0.25f
#define IOU_T   0.45f
#define MAX_WH  4096.0f
#define NMS_T   1024
#define DET_PAD 304   // 19 * 16 WMMA row tiles

typedef __attribute__((ext_vector_type(16))) _Float16 v16h;
typedef __attribute__((ext_vector_type(8)))  float    v8f;

// ---------------------------------------------------------------------------
// Kernel 1: decode. One wave32 per prediction: lanes cooperatively argmax the
// 80 class scores (coalesced lane loads + 5-step shfl_xor reduction), lane 0
// writes xyxy box, class-offset box (f32, exactly like reference), score, cls.
// ---------------------------------------------------------------------------
__global__ __launch_bounds__(256) void decode_kernel(
    const float* __restrict__ pred,
    float4* __restrict__ bo4, float4* __restrict__ bx4,
    float* __restrict__ sc,  float* __restrict__ cj)
{
  const int wid  = (blockIdx.x << 3) + (threadIdx.x >> 5);
  const int lane = threadIdx.x & 31;
  if (wid >= BDIM * NPRED) return;
  const float* p = pred + (size_t)wid * (5 + NCLS);

  // per-lane argmax over classes {lane, lane+32, lane+64(<80)}
  float bv = p[5 + lane];
  int   bi = lane;
  {
    float v1 = p[5 + 32 + lane];
    if (v1 > bv) { bv = v1; bi = lane + 32; }
    if (lane < 16) {
      float v2 = p[5 + 64 + lane];
      if (v2 > bv) { bv = v2; bi = lane + 64; }
    }
  }
  #pragma unroll
  for (int m = 16; m >= 1; m >>= 1) {          // wave32: 5 steps
    float ov = __shfl_xor(bv, m, 32);
    int   oi = __shfl_xor(bi, m, 32);
    if (ov > bv || (ov == bv && oi < bi)) { bv = ov; bi = oi; }
  }

  if (lane == 0) {
    float cx = p[0], cy = p[1], w = p[2], h = p[3], obj = p[4];
    float conf = bv * obj;                      // obj>=0 -> max commutes
    float x1 = cx - w * 0.5f, y1 = cy - h * 0.5f;
    float x2 = cx + w * 0.5f, y2 = cy + h * 0.5f;
    bool  valid = (obj > CONF_T) && (conf > CONF_T);
    float jf  = (float)bi;
    float off = jf * MAX_WH;                    // exact in f32
    bo4[wid] = make_float4(x1 + off, y1 + off, x2 + off, y2 + off); // b128 store
    bx4[wid] = make_float4(x1, y1, x2, y2);                         // b128 store
    sc[wid] = valid ? conf : -1.0f;
    cj[wid] = jf;
  }
}

// ---------------------------------------------------------------------------
// Kernel 2: greedy NMS. One workgroup (1024 thr = 32 wave32) per image.
// Scores live in LDS (101 KB -- fits the 320 KB WGP LDS). 300 iterations of
// block-wide argmax (shfl_xor within waves, LDS across waves) + IoU suppress
// against class-offset boxes streamed as float4 (global_load_b128) from L2
// (6.45 MB of boxes stay resident in the 192 MB L2). Writes dets/keep and the
// selection tables for the WMMA logits-gather kernel.
// ---------------------------------------------------------------------------
__global__ __launch_bounds__(NMS_T) void nms_kernel(
    const float4* __restrict__ bo4, const float4* __restrict__ bx4,
    const float* __restrict__ sc,  const float* __restrict__ cj,
    int* __restrict__ bidx_ws, float* __restrict__ bkeep_ws,
    float* __restrict__ dets, float* __restrict__ keep_out)
{
  extern __shared__ float sm[];
  float* s     = sm;                                   // NPRED scores
  float* bval  = sm + NPRED;                           // DET_PAD
  float* bkeep = bval + DET_PAD;                       // DET_PAD
  int*   bidx  = (int*)(bkeep + DET_PAD);              // DET_PAD
  float* redv  = (float*)(bidx + DET_PAD);             // 32
  int*   redi  = (int*)(redv + 32);                    // 32
  float* bcast = (float*)(redi + 32);                  // 2

  const int tid  = threadIdx.x;
  const int b    = blockIdx.x;
  const int lane = tid & 31, wv = tid >> 5;
  const float4* bo = bo4 + (size_t)b * NPRED;

  for (int i = tid; i < NPRED; i += NMS_T) s[i] = sc[(size_t)b * NPRED + i];
  for (int m = tid; m < DET_PAD; m += NMS_T) { bval[m] = 0.f; bkeep[m] = 0.f; bidx[m] = 0; }
  __syncthreads();

  for (int it = 0; it < MAX_DET; ++it) {
    // thread-local argmax over strided chunk (strict > keeps lowest index)
    float bv = s[tid];
    int   bi = tid;
    for (int k = 1; k < 25; ++k) {
      int i = tid + (k << 10);
      if (i < NPRED) { float v = s[i]; if (v > bv) { bv = v; bi = i; } }
    }
    #pragma unroll
    for (int m = 16; m >= 1; m >>= 1) {
      float ov = __shfl_xor(bv, m, 32);
      int   oi = __shfl_xor(bi, m, 32);
      if (ov > bv || (ov == bv && oi < bi)) { bv = ov; bi = oi; }
    }
    if (lane == 0) { redv[wv] = bv; redi[wv] = bi; }
    __syncthreads();
    if (wv == 0) {
      bv = redv[lane]; bi = redi[lane];
      #pragma unroll
      for (int m = 16; m >= 1; m >>= 1) {
        float ov = __shfl_xor(bv, m, 32);
        int   oi = __shfl_xor(bi, m, 32);
        if (ov > bv || (ov == bv && oi < bi)) { bv = ov; bi = oi; }
      }
      if (lane == 0) { bcast[0] = bv; ((int*)bcast)[1] = bi; }
    }
    __syncthreads();
    const float bestv = bcast[0];
    const int   besti = ((const int*)bcast)[1];
    if (bestv <= 0.0f) break;                 // uniform; remaining rows stay 0
    if (tid == 0) { bval[it] = bestv; bkeep[it] = 1.f; bidx[it] = besti; }

    // every thread loads the best box (single L2 line, broadcast)
    const float4 bb = bo[besti];
    const float a1 = (bb.z - bb.x) * (bb.w - bb.y);
    for (int k = 0; k < 25; ++k) {
      int i = tid + (k << 10);
      if (i < NPRED) {
        const float4 cb = bo[i];                        // global_load_b128
        float lx = fmaxf(bb.x, cb.x), ly = fmaxf(bb.y, cb.y);
        float rx = fminf(bb.z, cb.z), ry = fminf(bb.w, cb.w);
        float w = fmaxf(rx - lx, 0.f), h = fmaxf(ry - ly, 0.f);
        float inter = w * h;
        float a2 = (cb.z - cb.x) * (cb.w - cb.y);
        float iou = inter / (a1 + a2 - inter + 1e-9f);
        if (iou > IOU_T) s[i] = -1.0f;        // also suppresses best (IoU=1)
      }
    }
    __syncthreads();
  }
  __syncthreads();

  // export selection tables for the WMMA gather kernel
  for (int m = tid; m < DET_PAD; m += NMS_T) {
    bidx_ws[b * DET_PAD + m]  = bidx[m];
    bkeep_ws[b * DET_PAD + m] = bkeep[m];
  }
  // dets (precision-critical, 6 cols -> VALU) and keep
  const float4* bx  = bx4 + (size_t)b * NPRED;
  const float*  cjb = cj + (size_t)b * NPRED;
  for (int m = tid; m < MAX_DET; m += NMS_T) {
    float km = bkeep[m]; int id = bidx[m];
    const float4 bp = bx[id];                           // global_load_b128
    float* o = dets + ((size_t)b * MAX_DET + m) * 6;
    o[0] = bp.x * km; o[1] = bp.y * km; o[2] = bp.z * km; o[3] = bp.w * km;
    o[4] = bval[m] * km; o[5] = cjb[id] * km;
    keep_out[(size_t)b * MAX_DET + m] = km;
  }
}

// ---------------------------------------------------------------------------
// Kernel 3: lg[b,m,:] = logits[b, idx[m], :] * keep[m], computed on the WMMA
// pipe as D = diag(keep16) x L(16x16) with the exact hi/lo f16 split
// (L = f16(L) + f16(L - f16(L)); A is an exact 0/1 diagonal so D is
// f32-accurate). One wave32 per 16x16 tile; 19x5 tiles per image.
// ---------------------------------------------------------------------------
__global__ __launch_bounds__(256) void gather_logits_wmma_kernel(
    const float* __restrict__ logits, const int* __restrict__ bidx_ws,
    const float* __restrict__ bkeep_ws, float* __restrict__ lg)
{
  const int b    = blockIdx.x / 12;
  const int blkT = blockIdx.x % 12;
  const int wv   = threadIdx.x >> 5, lane = threadIdx.x & 31;
  const int t    = blkT * 8 + wv;
  if (t >= 95) return;                        // uniform per wave (EXEC stays full)
  const int tr = t / 5, tc = t % 5;
  const int col = tc * 16 + (lane & 15);

  const int*   bix = bidx_ws + b * DET_PAD;
  const float* bkp = bkeep_ws + b * DET_PAD;

  // A (16x32 f16) = diag(keep): lanes 0-15 hold K=0..15 (elem=lane);
  // everything else zero.
  const int dpos = (lane < 16) ? lane : 99;
  const int mrow = tr * 16 + (lane & 15);
  const _Float16 kh = (_Float16)bkp[mrow];

  v16h a, bh, bl;
  #pragma unroll
  for (int e = 0; e < 16; ++e) {
    a[e]  = (e == dpos) ? kh : (_Float16)0.0f;
    bh[e] = (_Float16)0.0f;
    bl[e] = (_Float16)0.0f;
  }
  // B (32x16 f16): lanes 0-15 carry rows K=0..15 (the data), lanes 16-31 zero.
  // Lanes 0-15 load consecutive columns of each gathered row -> coalesced 64B.
  if (lane < 16) {
    #pragma unroll
    for (int e = 0; e < 16; ++e) {
      int m = tr * 16 + e;                              // padded rows use idx 0, keep 0
      float v = logits[((size_t)b * NPRED + bix[m]) * NCLS + col];
      _Float16 h = (_Float16)v;
      bh[e] = h;
      bl[e] = (_Float16)(v - (float)h);                 // residual for exactness
    }
  }
  v8f c;
  #pragma unroll
  for (int r = 0; r < 8; ++r) c[r] = 0.0f;
  c = __builtin_amdgcn_wmma_f32_16x16x32_f16(false, a, false, bh, (short)0, c, false, false);
  c = __builtin_amdgcn_wmma_f32_16x16x32_f16(false, a, false, bl, (short)0, c, false, false);

  const int mbase = tr * 16 + ((lane < 16) ? 0 : 8);
  #pragma unroll
  for (int r = 0; r < 8; ++r) {
    int m2 = mbase + r;
    if (m2 < MAX_DET)
      lg[((size_t)b * MAX_DET + m2) * NCLS + col] = c[r];
  }
}

// ---------------------------------------------------------------------------
extern "C" void kernel_launch(void* const* d_in, const int* in_sizes, int n_in,
                              void* d_out, int out_size, void* d_ws, size_t ws_size,
                              hipStream_t stream)
{
  const float* pred   = (const float*)d_in[0];
  const float* logits = (const float*)d_in[1];

  float* out  = (float*)d_out;                       // tuple order: dets, lg, keep
  float* dets = out;
  float* lg   = out + (size_t)BDIM * MAX_DET * 6;
  float* keep = lg  + (size_t)BDIM * MAX_DET * NCLS;

  float* ws       = (float*)d_ws;
  float4* bo4     = (float4*)ws;                             // [B,N] offset boxes
  float4* bx4     = bo4 + (size_t)BDIM * NPRED;              // [B,N] raw boxes
  float* sc       = (float*)(bx4 + (size_t)BDIM * NPRED);    // [B,N] scores
  float* cjw      = sc + (size_t)BDIM * NPRED;               // [B,N] class idx (f32)
  int*   bidx_ws  = (int*)(cjw + (size_t)BDIM * NPRED);
  float* bkeep_ws = (float*)(bidx_ws + BDIM * DET_PAD);

  // Kernel 1: wave-per-prediction decode (bandwidth bound: ~137 MB read ~ 6us)
  decode_kernel<<<(BDIM * NPRED) / 8, 256, 0, stream>>>(pred, bo4, bx4, sc, cjw);

  // Kernel 2: per-image NMS, scores in LDS (~102 KB < 320 KB WGP LDS)
  const size_t smem = (size_t)(NPRED + 3 * DET_PAD + 32 + 32 + 2) * sizeof(float);
  (void)hipFuncSetAttribute((const void*)nms_kernel,
                            hipFuncAttributeMaxDynamicSharedMemorySize, (int)smem);
  nms_kernel<<<BDIM, NMS_T, smem, stream>>>(bo4, bx4, sc, cjw,
                                            bidx_ws, bkeep_ws, dets, keep);

  // Kernel 3: WMMA masked row-gather of logits (16 imgs x 95 tiles)
  gather_logits_wmma_kernel<<<BDIM * 12, 256, 0, stream>>>(logits, bidx_ws,
                                                           bkeep_ws, lg);
}
